// MyLSTM_57088705298740
// MI455X (gfx1250) — matrix operand order repaired and checked by reading, not compiled, not run on hardware
//
#include <hip/hip_runtime.h>

typedef __bf16 bf16_t;
typedef bf16_t v16bf __attribute__((ext_vector_type(16)));
typedef float  v8f   __attribute__((ext_vector_type(8)));
typedef unsigned int u32x4 __attribute__((ext_vector_type(4)));
typedef int          i32x8 __attribute__((ext_vector_type(8)));
typedef int          i32x4 __attribute__((ext_vector_type(4)));

#define B_SZ  128
#define T_SZ  256
#define INF   1024
#define CELLN 1024
#define OUTF  1024
#define KGATE 2048   // CELL + IN_F, concat order [cell | x]

// LDS x buffer: TDM-padded rows. 1024 elems/row + 16B pad after each 1024B
// written (pad_interval=256 DW, pad_amount=4 DW) -> row stride 2080 B (1040 elems),
// mid-row 16B gap after element 512. 2080 B = 520 DW == 8 (mod 64 banks).
#define XROW_ELEMS 1040

union VecB { v16bf v; uint4 q[2]; };

__device__ __forceinline__ unsigned short f2bf(float f) {
    unsigned int u = __float_as_uint(f);
    u += 0x7FFFu + ((u >> 16) & 1u);      // round-to-nearest-even
    return (unsigned short)(u >> 16);
}
__device__ __forceinline__ float sigmoidf_(float x) { return 1.0f / (1.0f + __expf(-x)); }
__device__ __forceinline__ float tanhf_(float x)    { return 1.0f - 2.0f / (__expf(2.0f * x) + 1.0f); }

// ---------------- one-time packing kernels ----------------

// Pack W[K][1024] (fp32, row-major) into bf16 WMMA-B tiles:
// group g = ((ntile*(K/32) + kc)*32 + lane), 16 contiguous bf16 per lane.
__global__ void pack_w(const float* __restrict__ src, unsigned short* __restrict__ dst, int K) {
    int g = blockIdx.x * blockDim.x + threadIdx.x;
    int kcN = K >> 5;
    int nGroups = 64 * kcN * 32;
    if (g >= nGroups) return;
    int lane  = g & 31;
    int rest  = g >> 5;
    int kc    = rest % kcN;
    int ntile = rest / kcN;
    int col   = ntile * 16 + (lane & 15);
    int kbase = kc * 32 + (lane >> 4) * 8;
    unsigned short* o = dst + (size_t)g * 16;
#pragma unroll
    for (int e = 0; e < 16; ++e)
        o[e] = f2bf(src[(size_t)(kbase + e + (e >= 8 ? 8 : 0)) * 1024 + col]);
}

// Convert x fp32 -> bf16 (same [B][T][1024] layout), 8 elems/thread
__global__ void conv_x(const float* __restrict__ src, unsigned short* __restrict__ dst) {
    size_t i = ((size_t)blockIdx.x * blockDim.x + threadIdx.x) * 8;
    const float4* s = (const float4*)(src + i);
    float4 a = s[0], b = s[1];
    unsigned short tmp[8];
    tmp[0] = f2bf(a.x); tmp[1] = f2bf(a.y); tmp[2] = f2bf(a.z); tmp[3] = f2bf(a.w);
    tmp[4] = f2bf(b.x); tmp[5] = f2bf(b.y); tmp[6] = f2bf(b.z); tmp[7] = f2bf(b.w);
    *(uint4*)(dst + i) = *(const uint4*)tmp;
}

// ---------------- TDM descriptor issue (whole-wave op, wave 0 only) ----------------
// 2D tile: 16 rows x 1024 bf16, row stride T*1024 elems, LDS padding enabled.
__device__ __forceinline__ void tdm_load_x(const unsigned short* __restrict__ xbf,
                                           int b0, int t, unsigned lds_off) {
    unsigned long long ga =
        (unsigned long long)(uintptr_t)(xbf + ((size_t)b0 * T_SZ + t) * (size_t)INF);
    u32x4 g0;
    g0[0] = 1u;                                        // count=1 (valid), user mode
    g0[1] = lds_off;                                   // lds_addr (bytes)
    g0[2] = (unsigned)(ga & 0xFFFFFFFFu);              // global_addr[31:0]
    g0[3] = (unsigned)((ga >> 32) & 0x01FFFFFFu)       // global_addr[56:32]
          | (2u << 30);                                // type = 2 ("image")
    i32x8 g1;
    g1[0] = (int)((1u << 16)      // data_size = 1 (2 bytes)
                | (1u << 20)      // pad_enable
                | (7u << 22)      // pad_interval: 256 DWORDs
                | (3u << 25));    // pad_amount: 4 DWORDs (16B)
    g1[1] = (int)(1024u << 16);   // tensor_dim0 = 1024 (low 16 in [31:16])
    g1[2] = (int)(16u << 16);     // tensor_dim0 hi=0, tensor_dim1 = 16 (low 16)
    g1[3] = (int)(1024u << 16);   // tensor_dim1 hi=0, tile_dim0 = 1024
    g1[4] = 16;                   // tile_dim1 = 16, tile_dim2 = 0
    g1[5] = (int)(T_SZ * INF);    // tensor_dim0_stride low32 (data_size units)
    g1[6] = 0;                    // stride hi, dim1_stride lo
    g1[7] = 0;
    i32x4 z4 = { 0, 0, 0, 0 };
    i32x8 z8 = { 0, 0, 0, 0, 0, 0, 0, 0 };
    __builtin_amdgcn_tensor_load_to_lds(g0, g1, z4, z4, z8, 0);
}

// ---------------- persistent LSTM kernel ----------------
__global__ __launch_bounds__(1024) void lstm_persistent(
    const unsigned short* __restrict__ xbf,
    const float* __restrict__ bfp, const float* __restrict__ bi1p,
    const float* __restrict__ bi2p, const float* __restrict__ bop,
    const float* __restrict__ bcp,
    const unsigned short* __restrict__ pWf,  const unsigned short* __restrict__ pWi1,
    const unsigned short* __restrict__ pWi2, const unsigned short* __restrict__ pWo,
    const unsigned short* __restrict__ pWc,
    float* __restrict__ outg, float* __restrict__ cellg)
{
    __shared__ unsigned short cellbuf[16][CELLN + 8];        // bf16 cell (A operand, K 0..1023)
    __shared__ unsigned short hbuf[16][CELLN + 8];           // bf16 h
    __shared__ unsigned short xraw[2][16 * XROW_ELEMS];      // TDM double buffer (K 1024..2047)

    const int tid   = threadIdx.x;
    const int lane  = tid & 31;
    const int w     = tid >> 5;        // wave 0..31, owns cols [32w, 32w+32)
    const int b0    = blockIdx.x * 16; // batch chunk
    const int mrow  = lane & 15;       // A-frag source row
    const int khalf = lane >> 4;       // A-frag K-half select

    const int ntv[2]  = { 2 * w, 2 * w + 1 };
    const int colv[2] = { ntv[0] * 16 + (lane & 15), ntv[1] * 16 + (lane & 15) };

    float bfv[2], bi1v[2], bi2v[2], bov[2], bcv[2];
#pragma unroll
    for (int j = 0; j < 2; ++j) {
        bfv[j]  = bfp[colv[j]];  bi1v[j] = bi1p[colv[j]];
        bi2v[j] = bi2p[colv[j]]; bov[j]  = bop[colv[j]];
        bcv[j]  = bcp[colv[j]];
    }

    // fp32 cell state in registers, WMMA C/D layout
    v8f cellr[2];
#pragma unroll
    for (int j = 0; j < 2; ++j)
#pragma unroll
        for (int v = 0; v < 8; ++v) cellr[j][v] = 0.0f;

    // zero bf16 cell region (each wave: its 32 cols x 16 rows)
#pragma unroll
    for (int v = 0; v < 8; ++v) {
        int m = v + 8 * khalf;
        cellbuf[m][colv[0]] = 0;
        cellbuf[m][colv[1]] = 0;
    }

    const unsigned xlds[2] = {
        (unsigned)(uintptr_t)(void*)&xraw[0][0],
        (unsigned)(uintptr_t)(void*)&xraw[1][0]
    };

    // prologue: async-load x_0 into buffer 0
    if (w == 0) tdm_load_x(xbf, b0, 0, xlds[0]);

    for (int t = 0; t < T_SZ; ++t) {
        const int cur = t & 1;

        if (w == 0) {
            if (t + 1 < T_SZ) {
                tdm_load_x(xbf, b0, t + 1, xlds[cur ^ 1]);   // prefetch next step
                __builtin_amdgcn_s_wait_tensorcnt(1);        // current buffer complete
            } else {
                __builtin_amdgcn_s_wait_tensorcnt(0);
            }
        }
        __syncthreads();   // xbuf[cur] + cell_bf16 visible to all waves

        // ---- gate GEMMs, C initialized with broadcast bias
        v8f af[2], a1[2], a2[2], ao_[2];
#pragma unroll
        for (int j = 0; j < 2; ++j)
#pragma unroll
            for (int v = 0; v < 8; ++v) {
                af[j][v] = bfv[j]; a1[j][v] = bi1v[j];
                a2[j][v] = bi2v[j]; ao_[j][v] = bov[j];
            }

        // K chunks 0..31: cell half of the concat
        for (int kc = 0; kc < CELLN / 32; ++kc) {
            VecB A;
            A.q[0] = *(const uint4*)&cellbuf[mrow][kc * 32 + khalf * 8];
            A.q[1] = *(const uint4*)&cellbuf[mrow][kc * 32 + 16 + khalf * 8];
            VecB Bb[8];
#pragma unroll
            for (int j = 0; j < 2; ++j) {
                size_t boff = ((size_t)(ntv[j] * (KGATE / 32) + kc) * 32 + lane) * 16;
                Bb[4*j+0].q[0] = *(const uint4*)&pWf[boff];   Bb[4*j+0].q[1] = *(const uint4*)&pWf[boff + 8];
                Bb[4*j+1].q[0] = *(const uint4*)&pWi1[boff];  Bb[4*j+1].q[1] = *(const uint4*)&pWi1[boff + 8];
                Bb[4*j+2].q[0] = *(const uint4*)&pWi2[boff];  Bb[4*j+2].q[1] = *(const uint4*)&pWi2[boff + 8];
                Bb[4*j+3].q[0] = *(const uint4*)&pWo[boff];   Bb[4*j+3].q[1] = *(const uint4*)&pWo[boff + 8];
            }
#pragma unroll
            for (int j = 0; j < 2; ++j) {
                af[j]  = __builtin_amdgcn_wmma_f32_16x16x32_bf16(false, A.v, false, Bb[4*j+0].v, (short)0, af[j],  false, false);
                a1[j]  = __builtin_amdgcn_wmma_f32_16x16x32_bf16(false, A.v, false, Bb[4*j+1].v, (short)0, a1[j],  false, false);
                a2[j]  = __builtin_amdgcn_wmma_f32_16x16x32_bf16(false, A.v, false, Bb[4*j+2].v, (short)0, a2[j],  false, false);
                ao_[j] = __builtin_amdgcn_wmma_f32_16x16x32_bf16(false, A.v, false, Bb[4*j+3].v, (short)0, ao_[j], false, false);
            }
        }
        // K chunks 32..63: x half of the concat (TDM-padded rows)
        {
            const char* xrow = (const char*)&xraw[cur][mrow * XROW_ELEMS];
            for (int kc = 0; kc < INF / 32; ++kc) {
                int xo1 = kc * 32 + khalf * 8;
                int xo2 = xo1 + 16;
                VecB A;
                A.q[0] = *(const uint4*)(xrow + xo1 * 2 + ((xo1 >> 9) << 4));
                A.q[1] = *(const uint4*)(xrow + xo2 * 2 + ((xo2 >> 9) << 4));
                int kcg = (CELLN / 32) + kc;
                VecB Bb[8];
#pragma unroll
                for (int j = 0; j < 2; ++j) {
                    size_t boff = ((size_t)(ntv[j] * (KGATE / 32) + kcg) * 32 + lane) * 16;
                    Bb[4*j+0].q[0] = *(const uint4*)&pWf[boff];   Bb[4*j+0].q[1] = *(const uint4*)&pWf[boff + 8];
                    Bb[4*j+1].q[0] = *(const uint4*)&pWi1[boff];  Bb[4*j+1].q[1] = *(const uint4*)&pWi1[boff + 8];
                    Bb[4*j+2].q[0] = *(const uint4*)&pWi2[boff];  Bb[4*j+2].q[1] = *(const uint4*)&pWi2[boff + 8];
                    Bb[4*j+3].q[0] = *(const uint4*)&pWo[boff];   Bb[4*j+3].q[1] = *(const uint4*)&pWo[boff + 8];
                }
#pragma unroll
                for (int j = 0; j < 2; ++j) {
                    af[j]  = __builtin_amdgcn_wmma_f32_16x16x32_bf16(false, A.v, false, Bb[4*j+0].v, (short)0, af[j],  false, false);
                    a1[j]  = __builtin_amdgcn_wmma_f32_16x16x32_bf16(false, A.v, false, Bb[4*j+1].v, (short)0, a1[j],  false, false);
                    a2[j]  = __builtin_amdgcn_wmma_f32_16x16x32_bf16(false, A.v, false, Bb[4*j+2].v, (short)0, a2[j],  false, false);
                    ao_[j] = __builtin_amdgcn_wmma_f32_16x16x32_bf16(false, A.v, false, Bb[4*j+3].v, (short)0, ao_[j], false, false);
                }
            }
        }
        __syncthreads();   // all gate reads of cellbuf done

        // ---- elementwise; cell update in registers; publish bf16 cell + h
#pragma unroll
        for (int j = 0; j < 2; ++j) {
#pragma unroll
            for (int v = 0; v < 8; ++v) {
                float fg = sigmoidf_(af[j][v]);
                float i1 = sigmoidf_(a1[j][v]);
                float i2 = tanhf_  (a2[j][v]);
                float og = sigmoidf_(ao_[j][v]);
                float c  = cellr[j][v] * fg + i1 * i2;
                cellr[j][v] = c;
                float h = og * tanhf_(c);
                int m = v + 8 * khalf;
                cellbuf[m][colv[j]] = f2bf(c);
                hbuf[m][colv[j]]    = f2bf(h);
            }
        }
        __syncthreads();   // h + new cell visible

        // ---- classifier GEMM: out = h @ Wc + bc  (C init = bias)
        v8f acc[2];
#pragma unroll
        for (int j = 0; j < 2; ++j)
#pragma unroll
            for (int v = 0; v < 8; ++v) acc[j][v] = bcv[j];

        for (int kc = 0; kc < CELLN / 32; ++kc) {
            VecB A;
            A.q[0] = *(const uint4*)&hbuf[mrow][kc * 32 + khalf * 8];
            A.q[1] = *(const uint4*)&hbuf[mrow][kc * 32 + 16 + khalf * 8];
            VecB Bc0, Bc1;
            {
                size_t b0f = ((size_t)(ntv[0] * (CELLN / 32) + kc) * 32 + lane) * 16;
                size_t b1f = ((size_t)(ntv[1] * (CELLN / 32) + kc) * 32 + lane) * 16;
                Bc0.q[0] = *(const uint4*)&pWc[b0f]; Bc0.q[1] = *(const uint4*)&pWc[b0f + 8];
                Bc1.q[0] = *(const uint4*)&pWc[b1f]; Bc1.q[1] = *(const uint4*)&pWc[b1f + 8];
            }
            acc[0] = __builtin_amdgcn_wmma_f32_16x16x32_bf16(false, A.v, false, Bc0.v, (short)0, acc[0], false, false);
            acc[1] = __builtin_amdgcn_wmma_f32_16x16x32_bf16(false, A.v, false, Bc1.v, (short)0, acc[1], false, false);
        }
#pragma unroll
        for (int j = 0; j < 2; ++j) {
#pragma unroll
            for (int v = 0; v < 8; ++v) {
                int m = v + 8 * khalf;
                outg[((size_t)(b0 + m) * T_SZ + t) * OUTF + colv[j]] = acc[j][v];
            }
        }
    }

    // ---- final cell state [1, B, CELL]
#pragma unroll
    for (int j = 0; j < 2; ++j)
#pragma unroll
        for (int v = 0; v < 8; ++v) {
            int m = v + 8 * khalf;
            cellg[(size_t)(b0 + m) * CELLN + colv[j]] = cellr[j][v];
        }
}

extern "C" void kernel_launch(void* const* d_in, const int* in_sizes, int n_in,
                              void* d_out, int out_size, void* d_ws, size_t ws_size,
                              hipStream_t stream) {
    const float* x   = (const float*)d_in[0];
    const float* Wf  = (const float*)d_in[1];
    const float* bf_ = (const float*)d_in[2];
    const float* Wi1 = (const float*)d_in[3];
    const float* bi1 = (const float*)d_in[4];
    const float* Wi2 = (const float*)d_in[5];
    const float* bi2 = (const float*)d_in[6];
    const float* Wo  = (const float*)d_in[7];
    const float* bo_ = (const float*)d_in[8];
    const float* Wc  = (const float*)d_in[9];
    const float* bc_ = (const float*)d_in[10];

    // workspace: bf16 packed weights (18.9 MB) + bf16 x (67.1 MB)
    unsigned short* ws = (unsigned short*)d_ws;
    const size_t WGELEMS = (size_t)2048 * 1024;
    const size_t WCELEMS = (size_t)1024 * 1024;
    unsigned short* pWf  = ws;
    unsigned short* pWi1 = ws + 1 * WGELEMS;
    unsigned short* pWi2 = ws + 2 * WGELEMS;
    unsigned short* pWo  = ws + 3 * WGELEMS;
    unsigned short* pWc  = ws + 4 * WGELEMS;
    unsigned short* xbf  = ws + 4 * WGELEMS + WCELEMS;

    const int gGate = (64 * (2048 / 32) * 32) / 256;  // 512 blocks
    const int gCls  = (64 * (1024 / 32) * 32) / 256;  // 256 blocks
    pack_w<<<gGate, 256, 0, stream>>>(Wf,  pWf,  2048);
    pack_w<<<gGate, 256, 0, stream>>>(Wi1, pWi1, 2048);
    pack_w<<<gGate, 256, 0, stream>>>(Wi2, pWi2, 2048);
    pack_w<<<gGate, 256, 0, stream>>>(Wo,  pWo,  2048);
    pack_w<<<gCls,  256, 0, stream>>>(Wc,  pWc,  1024);

    const size_t xElems = (size_t)B_SZ * T_SZ * INF;          // 33.5M
    conv_x<<<(int)(xElems / 8 / 256), 256, 0, stream>>>(x, xbf);

    float* outg  = (float*)d_out;
    float* cellg = outg + (size_t)B_SZ * T_SZ * OUTF;
    lstm_persistent<<<B_SZ / 16, 1024, 0, stream>>>(
        xbf, bf_, bi1, bi2, bo_, bc_, pWf, pWi1, pWi2, pWo, pWc, outg, cellg);
}
